// G2PLSTM_57449482551466
// MI455X (gfx1250) — matrix operand order repaired and checked by reading, not compiled, not run on hardware
//
#include <hip/hip_runtime.h>
#include <hip/hip_bf16.h>
#include <stdint.h>

// ---------------------------------------------------------------------------
// Seq2seq LSTM (L=3, H=256, E=300, B=8192) on gfx1250 via bf16 WMMA.
// Layer-by-layer over time; per (layer, t) one fused kernel:
//   g = [x_t | h_{t-1}] @ [Wih | Whh]^T + b   (v_wmma_f32_16x16x32_bf16)
// then the LSTM pointwise entirely in registers (gate-aligned wave columns).
// Block = 32 batch rows x all 1024 gate columns (2 M-tiles per wave) so each
// B-fragment fetched from L2 feeds two WMMAs -> halves weight traffic vs M=16.
// ---------------------------------------------------------------------------

typedef __attribute__((ext_vector_type(16))) __bf16 bf16x16;
typedef __attribute__((ext_vector_type(8)))  float   f32x8;

#define LDIM   3
#define HDIM   256
#define BATCH  8192
#define G4H    1024
#define EDIM   300
#define EPAD   320
#define TSRC   32
#define TTRG   32
#define VOUTD  128
#define MROWS  32        // batch rows per block (2 x 16 M-tiles)

union Frag { uint32_t u[8]; bf16x16 v; };

// ---------------------------------------------------------------------------
// One LSTM step for one layer: 256 blocks x 256 threads (8 waves).
// Wave w covers columns {g*256 + w*32 + s*16 + nloc : g in 0..3, s in 0..1}
// for both M-tiles, so every lane holds the (i,f,g,o) tuple of its outputs.
// ---------------------------------------------------------------------------
template <bool IS_L0>
__global__ __launch_bounds__(256)
void lstm_step_kernel(const __bf16* __restrict__ Wcomb,   // [1024][KTOT] bf16
                      const float*  __restrict__ bias,    // [1024]
                      const float*  __restrict__ emb,     // [V][300]   (L0 only)
                      const int*    __restrict__ tokens,  // [B]        (L0 only)
                      const __bf16* __restrict__ seq_in,  // [B][H]     (L>0 only)
                      __bf16*       __restrict__ h_state, // [B][H]  in/out
                      float*        __restrict__ c_state, // [B][H]  in/out
                      __bf16*       __restrict__ seq_out) // [B][H]  out (t slice)
{
    constexpr int KTOT  = IS_L0 ? (EPAD + HDIM) : (2 * HDIM); // 576 / 512
    constexpr int KHOFF = IS_L0 ? EPAD : HDIM;

    __shared__ __bf16 ldsA[MROWS * KTOT];

    const int tid  = threadIdx.x;
    const int row0 = blockIdx.x * MROWS;

    // ---- Stage A tile (MROWS x KTOT) into LDS, converting/gathering ----
    for (int e = tid; e < MROWS * KTOT; e += 256) {
        const int r = e / KTOT;
        const int k = e - r * KTOT;
        const int b = row0 + r;
        __bf16 val;
        if (k >= KHOFF) {
            val = h_state[(size_t)b * HDIM + (k - KHOFF)];
        } else if (IS_L0) {
            const int tok = tokens[b];
            val = (k < EDIM) ? (__bf16)emb[(size_t)tok * EDIM + k] : (__bf16)0.0f;
        } else {
            val = seq_in[(size_t)b * HDIM + k];
        }
        ldsA[e] = val;
    }
    __syncthreads();

    const int lane = tid & 31;
    const int wave = tid >> 5;
    const int half = lane >> 4;   // 0/1
    const int nloc = lane & 15;
    const int mrow = lane & 15;

    const f32x8 vzero = {0.f, 0.f, 0.f, 0.f, 0.f, 0.f, 0.f, 0.f};
    f32x8 acc[2][8];
    #pragma unroll
    for (int mt = 0; mt < 2; ++mt)
        #pragma unroll
        for (int i = 0; i < 8; ++i) acc[mt][i] = vzero;

    #pragma unroll 1
    for (int kc = 0; kc < KTOT / 32; ++kc) {
        // A fragments (16x32 bf16 each): lane=M, VGPR v -> K pair per ISA.
        Frag a[2];
        #pragma unroll
        for (int mt = 0; mt < 2; ++mt) {
            const uint32_t* a32 =
                (const uint32_t*)(ldsA + (mt * 16 + mrow) * KTOT) + kc * 16;
            #pragma unroll
            for (int v = 0; v < 8; ++v)
                a[mt].u[v] = a32[(v >> 2) * 8 + half * 4 + (v & 3)];
        }

        #pragma unroll
        for (int nt = 0; nt < 8; ++nt) {
            const int gate = nt >> 1;
            const int sub  = nt & 1;
            const int ncol = gate * 256 + wave * 32 + sub * 16 + nloc;
            // B fragment (32x16 bf16): lane=N, VGPR v holds K = (v>>2)*16 + half*8 + 2*(v&3).
            const uint32_t* b32 =
                (const uint32_t*)(Wcomb + (size_t)ncol * KTOT) + kc * 16;
            Frag bb;
            #pragma unroll
            for (int v = 0; v < 8; ++v)
                bb.u[v] = b32[(v >> 2) * 8 + half * 4 + (v & 3)];

            #pragma unroll
            for (int mt = 0; mt < 2; ++mt)
                acc[mt][nt] = __builtin_amdgcn_wmma_f32_16x16x32_bf16(
                    false, a[mt].v, false, bb.v, (short)0, acc[mt][nt],
                    false, false);
        }
    }

    // ---- Pointwise LSTM in registers: acc[mt][gate*2+sub] share (m, j). ----
    #pragma unroll
    for (int mt = 0; mt < 2; ++mt) {
        #pragma unroll
        for (int sub = 0; sub < 2; ++sub) {
            const int j   = wave * 32 + sub * 16 + nloc;
            const float bi = bias[j];
            const float bf = bias[256 + j];
            const float bg = bias[512 + j];
            const float bo = bias[768 + j];
            const f32x8 gi = acc[mt][0 + sub];
            const f32x8 gf = acc[mt][2 + sub];
            const f32x8 gg = acc[mt][4 + sub];
            const f32x8 go = acc[mt][6 + sub];
            #pragma unroll
            for (int r = 0; r < 8; ++r) {
                const int    m   = mt * 16 + half * 8 + r;
                const size_t off = (size_t)(row0 + m) * HDIM + j;
                const float xi = gi[r] + bi;
                const float xf = gf[r] + bf;
                const float xg = gg[r] + bg;
                const float xo = go[r] + bo;
                const float si = 1.0f / (1.0f + __expf(-xi));
                const float sf = 1.0f / (1.0f + __expf(-xf));
                const float so = 1.0f / (1.0f + __expf(-xo));
                const float cn = sf * c_state[off] + si * tanhf(xg);
                const float hn = so * tanhf(cn);
                c_state[off] = cn;
                const __bf16 hb = (__bf16)hn;
                h_state[off] = hb;
                seq_out[off] = hb;
            }
        }
    }
}

// ---------------------------------------------------------------------------
// Final FC: logits = ys @ fc_W^T + fc_b, written to out rows [B, 32B).
// Block = 16 rows x 128 cols; wave w owns one 16x16 tile (cols w*16..).
// ---------------------------------------------------------------------------
__global__ __launch_bounds__(256)
void fc_kernel(const __bf16* __restrict__ seq,  // [31*B][H] bf16
               const __bf16* __restrict__ Wfc,  // [128][256] bf16
               const float*  __restrict__ bfc,  // [128]
               float*        __restrict__ out)  // [32*B][128]
{
    __shared__ __bf16 ldsA[16 * HDIM];
    const int tid  = threadIdx.x;
    const int row0 = blockIdx.x * 16;

    for (int e = tid; e < 16 * HDIM; e += 256) {
        const int r = e >> 8;
        const int k = e & 255;
        ldsA[e] = seq[(size_t)(row0 + r) * HDIM + k];
    }
    __syncthreads();

    const int lane = tid & 31;
    const int wave = tid >> 5;
    const int half = lane >> 4;
    const int nloc = lane & 15;
    const int mrow = lane & 15;

    f32x8 acc = {0.f, 0.f, 0.f, 0.f, 0.f, 0.f, 0.f, 0.f};
    const uint32_t* arow = (const uint32_t*)(ldsA + mrow * HDIM);
    const int ncol = wave * 16 + nloc;
    const uint32_t* brow = (const uint32_t*)(Wfc + (size_t)ncol * HDIM);

    #pragma unroll
    for (int kc = 0; kc < HDIM / 32; ++kc) {
        Frag a, bb;
        const uint32_t* a32 = arow + kc * 16;
        const uint32_t* b32 = brow + kc * 16;
        #pragma unroll
        for (int v = 0; v < 8; ++v) {
            a.u[v]  = a32[(v >> 2) * 8 + half * 4 + (v & 3)];
            bb.u[v] = b32[(v >> 2) * 8 + half * 4 + (v & 3)];
        }
        acc = __builtin_amdgcn_wmma_f32_16x16x32_bf16(
            false, a.v, false, bb.v, (short)0, acc, false, false);
    }

    const float bv = bfc[ncol];
    #pragma unroll
    for (int r = 0; r < 8; ++r) {
        const int    m    = half * 8 + r;
        const size_t grow = (size_t)(row0 + m);
        out[(grow + BATCH) * VOUTD + ncol] = acc[r] + bv;
    }
}

// ---------------------------------------------------------------------------
// Weight pre-pack: dst[n][k] = k < Kx ? (k < Ein ? Wih[n][k] : 0)
//                                     : Whh[n][k-Kx]   (bf16, K contiguous)
// ---------------------------------------------------------------------------
__global__ void pack_weights_kernel(const float* __restrict__ Wih,
                                    const float* __restrict__ Whh,
                                    __bf16* __restrict__ dst,
                                    int N, int Ein, int Kx, int Ktot)
{
    const long i = (long)blockIdx.x * 256 + threadIdx.x;
    if (i >= (long)N * Ktot) return;
    const int n = (int)(i / Ktot);
    const int k = (int)(i - (long)n * Ktot);
    float v;
    if (k < Kx) v = (k < Ein) ? Wih[(long)n * Ein + k] : 0.0f;
    else        v = Whh[(long)n * HDIM + (k - Kx)];
    dst[i] = (__bf16)v;
}

__global__ void zero_u32_kernel(uint32_t* __restrict__ p, long n)
{
    const long stride = (long)gridDim.x * blockDim.x;
    for (long i = (long)blockIdx.x * blockDim.x + threadIdx.x; i < n; i += stride)
        p[i] = 0u;
}

// ---------------------------------------------------------------------------
extern "C" void kernel_launch(void* const* d_in, const int* in_sizes, int n_in,
                              void* d_out, int out_size, void* d_ws, size_t ws_size,
                              hipStream_t stream)
{
    (void)in_sizes; (void)n_in; (void)out_size; (void)ws_size;

    const int*   src      = (const int*)  d_in[0];
    const int*   trg      = (const int*)  d_in[1];
    const float* enc_emb  = (const float*)d_in[2];
    const float* enc_Wih0 = (const float*)d_in[3];
    const float* enc_WihH = (const float*)d_in[4];
    const float* enc_Whh  = (const float*)d_in[5];
    const float* enc_b    = (const float*)d_in[6];
    const float* dec_emb  = (const float*)d_in[7];
    const float* dec_Wih0 = (const float*)d_in[8];
    const float* dec_WihH = (const float*)d_in[9];
    const float* dec_Whh  = (const float*)d_in[10];
    const float* dec_b    = (const float*)d_in[11];
    const float* fc_W     = (const float*)d_in[12];
    const float* fc_b     = (const float*)d_in[13];
    float* out = (float*)d_out;

    // ---- Workspace carve-out (all 256B aligned) ----
    char*  ws  = (char*)d_ws;
    size_t off = 0;
    auto alloc = [&](size_t bytes) -> char* {
        char* p = ws + off;
        off = (off + bytes + 255) & ~(size_t)255;
        return p;
    };
    const size_t K0 = EPAD + HDIM;  // 576
    const size_t K1 = 2 * HDIM;     // 512
    __bf16* Wenc0 = (__bf16*)alloc(G4H * K0 * 2);
    __bf16* Wenc1 = (__bf16*)alloc(G4H * K1 * 2);
    __bf16* Wenc2 = (__bf16*)alloc(G4H * K1 * 2);
    __bf16* Wdec0 = (__bf16*)alloc(G4H * K0 * 2);
    __bf16* Wdec1 = (__bf16*)alloc(G4H * K1 * 2);
    __bf16* Wdec2 = (__bf16*)alloc(G4H * K1 * 2);
    __bf16* Wfc   = (__bf16*)alloc((size_t)VOUTD * HDIM * 2);
    __bf16* hstate = (__bf16*)alloc((size_t)LDIM * BATCH * HDIM * 2);
    float*  cstate = (float*) alloc((size_t)LDIM * BATCH * HDIM * 4);
    __bf16* seqA   = (__bf16*)alloc((size_t)TSRC * BATCH * HDIM * 2);
    __bf16* seqB   = (__bf16*)alloc((size_t)TSRC * BATCH * HDIM * 2);

    auto cdiv = [](long a, long b) { return (int)((a + b - 1) / b); };
    const size_t SL = (size_t)BATCH * HDIM;   // one (t, layer) state slice

    // ---- Pack weights to bf16 [Wih | Whh], K contiguous ----
    pack_weights_kernel<<<cdiv((long)G4H * K0, 256), 256, 0, stream>>>(
        enc_Wih0, enc_Whh, Wenc0, G4H, EDIM, EPAD, (int)K0);
    pack_weights_kernel<<<cdiv((long)G4H * K1, 256), 256, 0, stream>>>(
        enc_WihH, enc_Whh + (size_t)1 * G4H * HDIM, Wenc1, G4H, HDIM, HDIM, (int)K1);
    pack_weights_kernel<<<cdiv((long)G4H * K1, 256), 256, 0, stream>>>(
        enc_WihH + (size_t)G4H * HDIM, enc_Whh + (size_t)2 * G4H * HDIM,
        Wenc2, G4H, HDIM, HDIM, (int)K1);
    pack_weights_kernel<<<cdiv((long)G4H * K0, 256), 256, 0, stream>>>(
        dec_Wih0, dec_Whh, Wdec0, G4H, EDIM, EPAD, (int)K0);
    pack_weights_kernel<<<cdiv((long)G4H * K1, 256), 256, 0, stream>>>(
        dec_WihH, dec_Whh + (size_t)1 * G4H * HDIM, Wdec1, G4H, HDIM, HDIM, (int)K1);
    pack_weights_kernel<<<cdiv((long)G4H * K1, 256), 256, 0, stream>>>(
        dec_WihH + (size_t)G4H * HDIM, dec_Whh + (size_t)2 * G4H * HDIM,
        Wdec2, G4H, HDIM, HDIM, (int)K1);
    pack_weights_kernel<<<cdiv((long)VOUTD * HDIM, 256), 256, 0, stream>>>(
        fc_W, nullptr, Wfc, VOUTD, HDIM, HDIM, HDIM);

    // ---- Zero initial states and the t=0 output slab ----
    zero_u32_kernel<<<2048, 256, 0, stream>>>((uint32_t*)hstate, (long)LDIM * SL / 2);
    zero_u32_kernel<<<2048, 256, 0, stream>>>((uint32_t*)cstate, (long)LDIM * SL);
    zero_u32_kernel<<<2048, 256, 0, stream>>>((uint32_t*)out, (long)BATCH * VOUTD);

    const int grid = BATCH / MROWS;  // 256 blocks per step

    // ---- Encoder: layer-by-layer over 32 steps ----
    for (int t = 0; t < TSRC; ++t)
        lstm_step_kernel<true><<<grid, 256, 0, stream>>>(
            Wenc0, enc_b, enc_emb, src + (size_t)t * BATCH, nullptr,
            hstate, cstate, seqA + (size_t)t * SL);
    for (int t = 0; t < TSRC; ++t)
        lstm_step_kernel<false><<<grid, 256, 0, stream>>>(
            Wenc1, enc_b + G4H, nullptr, nullptr, seqA + (size_t)t * SL,
            hstate + SL, cstate + SL, seqB + (size_t)t * SL);
    for (int t = 0; t < TSRC; ++t)
        lstm_step_kernel<false><<<grid, 256, 0, stream>>>(
            Wenc2, enc_b + 2 * G4H, nullptr, nullptr, seqB + (size_t)t * SL,
            hstate + 2 * SL, cstate + 2 * SL, seqA + (size_t)t * SL);

    // ---- Decoder: states carry over from encoder; 31 steps ----
    const int TD = TTRG - 1;
    for (int t = 0; t < TD; ++t)
        lstm_step_kernel<true><<<grid, 256, 0, stream>>>(
            Wdec0, dec_b, dec_emb, trg + (size_t)t * BATCH, nullptr,
            hstate, cstate, seqA + (size_t)t * SL);
    for (int t = 0; t < TD; ++t)
        lstm_step_kernel<false><<<grid, 256, 0, stream>>>(
            Wdec1, dec_b + G4H, nullptr, nullptr, seqA + (size_t)t * SL,
            hstate + SL, cstate + SL, seqB + (size_t)t * SL);
    for (int t = 0; t < TD; ++t)
        lstm_step_kernel<false><<<grid, 256, 0, stream>>>(
            Wdec2, dec_b + 2 * G4H, nullptr, nullptr, seqB + (size_t)t * SL,
            hstate + 2 * SL, cstate + 2 * SL, seqA + (size_t)t * SL);

    // ---- Final FC over all 31 decoder outputs ----
    fc_kernel<<<(TD * BATCH) / 16, 256, 0, stream>>>(seqA, Wfc, fc_b, out);
}
